// CUTLASSGroupedLinear_41601053229356
// MI455X (gfx1250) — matrix operand order repaired and checked by reading, not compiled, not run on hardware
//
#include <hip/hip_runtime.h>
#include <hip/hip_bf16.h>
#include <stdint.h>

// ---------------------------------------------------------------------------
// Grouped GEMM  Y_e = X_e @ W_e^T   (E=64, M=256/expert, K=N=2048, fp32 I/O)
// fp32 -> bf16 cooperative convert into LDS, V_WMMA_F32_16X16X32_BF16 with
// fp32 accumulators. 128x128 block tile, 8 wave32 each doing 32x64.
// Global loads for tile kt+BK are software-pipelined through registers so
// HBM latency is hidden under the WMMA + ds_load work of tile kt.
// ---------------------------------------------------------------------------

typedef __bf16 bf16_t;
typedef __attribute__((ext_vector_type(16))) __bf16 bf16x16;
typedef __attribute__((ext_vector_type(8)))  float  f32x8;

#define NUM_EXPERTS 64
#define K_DIM 2048
#define N_DIM 2048
#define TPE   256      // tokens per expert
#define BM    128
#define BN    128
#define BK    64
#define LDSS  72       // LDS row stride in bf16 elems (64 + 8 pad, 144 B, 16B-aligned)

union FragAB {
    bf16x16 v;
    uint4   q[2];
};

__global__ __launch_bounds__(256)
void grouped_gemm_bf16_wmma(const float* __restrict__ X,
                            const float* __restrict__ W,
                            const int*   __restrict__ m_sizes,
                            const int*   __restrict__ m_offsets,
                            float*       __restrict__ Y) {
    __shared__ __align__(16) bf16_t sX[BM * LDSS];
    __shared__ __align__(16) bf16_t sW[BN * LDSS];

    const int e  = blockIdx.z;
    const int m0 = m_offsets[e] + blockIdx.y * BM;   // global token row base
    const int n0 = blockIdx.x * BN;                  // output feature base
    const float* We = W + (size_t)e * N_DIM * K_DIM;

    const int tid   = threadIdx.x;
    const int lane  = tid & 31;
    const int wave  = tid >> 5;
    const int wm    = wave & 3;    // wave row block (x32) within 128
    const int wn    = wave >> 2;   // wave col block (x64) within 128
    const int lrow  = lane & 15;   // M-row (A) / N-col (B) held by this lane
    const int khalf = lane >> 4;   // which K-half of the fragment

    // staging map: each of 256 threads converts 32 fp32 of X and 32 of W
    const int sr = tid >> 1;          // tile row 0..127
    const int sh = (tid & 1) * 32;    // K sub-column 0 or 32

    const float* gX = X  + (size_t)(m0 + sr) * K_DIM + sh;
    const float* gW = We + (size_t)(n0 + sr) * K_DIM + sh;
    uint4* dX = (uint4*)&sX[sr * LDSS + sh];
    uint4* dW = (uint4*)&sW[sr * LDSS + sh];

    f32x8 c[2][4];
#pragma unroll
    for (int i = 0; i < 2; ++i)
#pragma unroll
        for (int j = 0; j < 4; ++j)
            c[i][j] = (f32x8)0.0f;

    // ---- prefetch tile kt=0 into registers ----
    float4 px[8], pw[8];
#pragma unroll
    for (int j = 0; j < 8; ++j) {
        px[j] = *(const float4*)(gX + 4 * j);
        pw[j] = *(const float4*)(gW + 4 * j);
    }

    for (int kt = 0; kt < K_DIM; kt += BK) {
        // ---- convert prefetched fp32 -> bf16, store to LDS ----
        {
            union { bf16_t b[32]; uint4 q[4]; } t;
#pragma unroll
            for (int j = 0; j < 8; ++j) {
                t.b[4 * j + 0] = (bf16_t)px[j].x;  t.b[4 * j + 1] = (bf16_t)px[j].y;
                t.b[4 * j + 2] = (bf16_t)px[j].z;  t.b[4 * j + 3] = (bf16_t)px[j].w;
            }
#pragma unroll
            for (int j = 0; j < 4; ++j) dX[j] = t.q[j];
#pragma unroll
            for (int j = 0; j < 8; ++j) {
                t.b[4 * j + 0] = (bf16_t)pw[j].x;  t.b[4 * j + 1] = (bf16_t)pw[j].y;
                t.b[4 * j + 2] = (bf16_t)pw[j].z;  t.b[4 * j + 3] = (bf16_t)pw[j].w;
            }
#pragma unroll
            for (int j = 0; j < 4; ++j) dW[j] = t.q[j];
        }
        __syncthreads();

        // ---- issue global loads for tile kt+BK NOW (uniform branch);
        //      latency hides under the WMMA + ds_load work below ----
        if (kt + BK < K_DIM) {
            const float* nX = gX + kt + BK;
            const float* nW = gW + kt + BK;
#pragma unroll
            for (int j = 0; j < 8; ++j) {
                px[j] = *(const float4*)(nX + 4 * j);
                pw[j] = *(const float4*)(nW + 4 * j);
            }
        }

        // ---- two K=32 WMMA steps per staged BK=64 ----
#pragma unroll
        for (int kk = 0; kk < BK; kk += 32) {
            FragAB a[2];
#pragma unroll
            for (int i = 0; i < 2; ++i) {
                // A 16x32 bf16 layout: lanes 0-15 K 0-7/16-23, lanes 16-31 K 8-15/24-31
                const bf16_t* p = &sX[(wm * 32 + i * 16 + lrow) * LDSS + kk + khalf * 8];
                a[i].q[0] = *(const uint4*)p;          // K  0.. 7 (+khalf*8)
                a[i].q[1] = *(const uint4*)(p + 16);   // K 16..23 (+khalf*8)
            }
            FragAB b[4];
#pragma unroll
            for (int j = 0; j < 4; ++j) {
                // B 32x16 bf16 layout: lanes 0-15 K 0-7/16-23, lanes 16-31 K 8-15/24-31
                const bf16_t* p = &sW[(wn * 64 + j * 16 + lrow) * LDSS + kk + khalf * 8];
                b[j].q[0] = *(const uint4*)p;          // K  0.. 7 (+khalf*8)
                b[j].q[1] = *(const uint4*)(p + 16);   // K 16..23 (+khalf*8)
            }
#pragma unroll
            for (int i = 0; i < 2; ++i)
#pragma unroll
                for (int j = 0; j < 4; ++j)
                    c[i][j] = __builtin_amdgcn_wmma_f32_16x16x32_bf16(
                        false, a[i].v, false, b[j].v,
                        (short)0, c[i][j], false, false);
        }
        __syncthreads();
    }

    // ---- epilogue: C/D layout -> global fp32 ----
#pragma unroll
    for (int i = 0; i < 2; ++i) {
#pragma unroll
        for (int j = 0; j < 4; ++j) {
            const int row = m0 + wm * 32 + i * 16 + khalf * 8;
            float* out = Y + (size_t)row * N_DIM + n0 + wn * 64 + j * 16 + lrow;
#pragma unroll
            for (int r = 0; r < 8; ++r)
                out[(size_t)r * N_DIM] = c[i][j][r];
        }
    }
}

extern "C" void kernel_launch(void* const* d_in, const int* in_sizes, int n_in,
                              void* d_out, int out_size, void* d_ws, size_t ws_size,
                              hipStream_t stream) {
    const float* X         = (const float*)d_in[0];
    const float* W         = (const float*)d_in[1];
    const int*   m_sizes   = (const int*)d_in[2];
    const int*   m_offsets = (const int*)d_in[3];
    float*       Y         = (float*)d_out;

    dim3 grid(N_DIM / BN, TPE / BM, NUM_EXPERTS);   // (16, 2, 64)
    dim3 block(256);
    hipLaunchKernelGGL(grouped_gemm_bf16_wmma, grid, block, 0, stream,
                       X, W, m_sizes, m_offsets, Y);
}
